// HomeObjectDetectionModel_10307921510472
// MI455X (gfx1250) — matrix operand order repaired and checked by reading, not compiled, not run on hardware
//
#include <hip/hip_runtime.h>

#define EPS 1e-5f

typedef __attribute__((ext_vector_type(16))) _Float16 v16h;
typedef __attribute__((ext_vector_type(8)))  float    v8f;
typedef __attribute__((ext_vector_type(4)))  int      v4i;

// ---- gfx1250 async global->LDS copy (ASYNCcnt path), guarded so absence of
// ---- the builtin cannot break the build.
#if defined(__has_builtin)
#if __has_builtin(__builtin_amdgcn_global_load_async_to_lds_b128) && \
    __has_builtin(__builtin_amdgcn_s_wait_asynccnt)
#define HAVE_ASYNC_LDS 1
#endif
#endif
#ifndef HAVE_ASYNC_LDS
#define HAVE_ASYNC_LDS 0
#endif

// ---------------------------------------------------------------------------
// f32 -> f16 weight conversion (run once per launch; tiny)
// ---------------------------------------------------------------------------
__global__ void f32_to_f16_kernel(const float* __restrict__ src,
                                  _Float16* __restrict__ dst, int n)
{
    int i = blockIdx.x * 256 + threadIdx.x;
    if (i < n) dst[i] = (_Float16)src[i];
}

// ---------------------------------------------------------------------------
// Stem: conv 3->32 (3x3, pad 1) + BN + ReLU6.  f32 input, f16 output.
// ---------------------------------------------------------------------------
__global__ void stem_conv_kernel(const float* __restrict__ X,
                                 const float* __restrict__ Wc,
                                 const float* __restrict__ g, const float* __restrict__ b,
                                 const float* __restrict__ m, const float* __restrict__ vv,
                                 _Float16* __restrict__ Out, int N, int H, int W)
{
    __shared__ float wsm[32 * 27];
    __shared__ float sc[32], sh[32];
    const int tl = threadIdx.x;
    for (int i = tl; i < 32 * 27; i += blockDim.x) wsm[i] = Wc[i];
    if (tl < 32) {
        float s = g[tl] * rsqrtf(vv[tl] + EPS);
        sc[tl] = s;
        sh[tl] = b[tl] - m[tl] * s;
    }
    __syncthreads();

    int t = blockIdx.x * blockDim.x + tl;
    int total = N * H * W;
    if (t >= total) return;
    int w = t % W;
    int r = t / W;
    int h = r % H;
    int n = r / H;

    float xv[27];
    #pragma unroll
    for (int ci = 0; ci < 3; ci++)
        #pragma unroll
        for (int dy = 0; dy < 3; dy++)
            #pragma unroll
            for (int dx = 0; dx < 3; dx++) {
                int hh = h + dy - 1, ww = w + dx - 1;
                bool ok = (hh >= 0 && hh < H && ww >= 0 && ww < W);
                int off = ((n * 3 + ci) * H + hh) * W + ww;
                float val = X[ok ? off : 0];
                xv[ci * 9 + dy * 3 + dx] = ok ? val : 0.f;
            }

    for (int co = 0; co < 32; co++) {
        float acc = 0.f;
        #pragma unroll
        for (int k = 0; k < 27; k++) acc = fmaf(wsm[co * 27 + k], xv[k], acc);
        acc = acc * sc[co] + sh[co];
        acc = fminf(fmaxf(acc, 0.f), 6.f);
        Out[((n * 32 + co) * H + h) * W + w] = (_Float16)acc;
    }
}

// ---------------------------------------------------------------------------
// Fused maxpool 2x2 + depthwise 3x3 (pad 1) + BN + ReLU6.  f16 in/out.
// ---------------------------------------------------------------------------
__global__ void pool_dw_kernel(const _Float16* __restrict__ X,
                               const float* __restrict__ dw,
                               const float* __restrict__ g, const float* __restrict__ b,
                               const float* __restrict__ m, const float* __restrict__ vv,
                               _Float16* __restrict__ Out, int N, int C, int Hin, int Win)
{
    const int Ho = Hin >> 1, Wo = Win >> 1;
    int t = blockIdx.x * blockDim.x + threadIdx.x;
    int total = N * C * Ho * Wo;
    if (t >= total) return;
    int wo = t % Wo;
    int r = t / Wo;
    int ho = r % Ho; r /= Ho;
    int c  = r % C;
    int n  = r / C;

    const _Float16* xin = X + (n * C + c) * Hin * Win;
    float acc = 0.f;
    #pragma unroll
    for (int i = 0; i < 3; i++) {
        int hp = ho + i - 1;
        #pragma unroll
        for (int j = 0; j < 3; j++) {
            int wp = wo + j - 1;
            float pv = 0.f;                      // conv zero-pad outside pooled map
            if (hp >= 0 && hp < Ho && wp >= 0 && wp < Wo) {
                const _Float16* r0 = xin + (hp * 2) * Win + wp * 2;
                float a0 = (float)r0[0],   a1 = (float)r0[1];
                float a2 = (float)r0[Win], a3 = (float)r0[Win + 1];
                pv = fmaxf(fmaxf(a0, a1), fmaxf(a2, a3));
            }
            acc = fmaf(dw[c * 9 + i * 3 + j], pv, acc);
        }
    }
    float s = g[c] * rsqrtf(vv[c] + EPS);
    acc = acc * s + (b[c] - m[c] * s);
    acc = fminf(fmaxf(acc, 0.f), 6.f);
    Out[((n * C + c) * Ho + ho) * Wo + wo] = (_Float16)acc;
}

// ---------------------------------------------------------------------------
// Implicit-GEMM conv via v_wmma_f32_16x16x32_f16, f16 operands, f32 accum.
//   Out[co,p] = sum_k W[co,k] * im2col(X)[k,p]     KH compile-time (1 or 3)
// Block 256 threads = 8 wave32, tile 64 co x 64 px, K step 32.
// A tile: async global->LDS b128 copies when in-bounds (ASYNCcnt), else
//         branch-free register staging.  pitch 40 f16 -> conflict-free.
// B tile: branch-free register-staged transpose (b64 global loads when HW%4==0).
// ---------------------------------------------------------------------------
template <int KH>
__global__ void conv_gemm_wmma_kernel(const _Float16* __restrict__ X,
                                      const _Float16* __restrict__ Wt,
                                      const float* __restrict__ g, const float* __restrict__ b,
                                      const float* __restrict__ m, const float* __restrict__ vv,
                                      const float* __restrict__ bias,
                                      float* __restrict__ OutF,         // f32 output (final)
                                      _Float16* __restrict__ OutH,      // f16 output (hidden)
                                      int N, int Cin, int H, int W, int Cout,
                                      int use_bn, int use_relu)
{
    __shared__ __align__(16) _Float16 lA[64 * 40];   // [co_row][k]  pitch 40
    __shared__ __align__(16) _Float16 lB[64 * 34];   // [px][k]      pitch 34

    const int HW = H * W;
    const int P = N * HW;
    const int Kdim = Cin * KH * KH;

    const int t    = threadIdx.x;
    const int lane = t & 31;
    const int wv   = t >> 5;
    const int coStrip = wv >> 1;
    const int pxStrip = wv & 1;
    const int pxBase = blockIdx.x * 64;
    const int coBase = blockIdx.y * 64;

    const bool aFull  = (coBase + 64) <= Cout;   // uniform
    const bool hwVec  = (HW & 3) == 0;           // uniform

    v8f acc0 = {};
    v8f acc1 = {};

    for (int kt = 0; kt < Kdim; kt += 32) {
        // ---------------- A tile: 64 co x 32 k of f16 weights ----------------
#if HAVE_ASYNC_LDS
        if (aFull) {
            // one async b128 per thread: row-contiguous in both global and LDS
            int row = t >> 2, chunk = t & 3;
            const _Float16* gp = Wt + (coBase + row) * Kdim + kt + chunk * 8;
            __builtin_amdgcn_global_load_async_to_lds_b128(
                (v4i*)gp, (v4i*)&lA[row * 40 + chunk * 8], 0, 0);
        } else
#endif
        {
            #pragma unroll
            for (int i = 0; i < 4; i++) {
                int e = t + i * 256;           // 1024 dwords
                int row = e >> 4;
                int ku  = (e & 15) * 2;
                int co  = coBase + row;
                bool ok = co < Cout;
                int off = co * Kdim + kt + ku;
                unsigned int val = *(const unsigned int*)(Wt + (ok ? off : 0));
                if (!ok) val = 0u;
                *(unsigned int*)&lA[row * 40 + ku] = val;
            }
        }

        // ---------------- B tile: 32 k x 64 px, stored px-major ----------------
        if (KH == 1 && hwVec) {
            // wide path: b64 global loads (4 px), branch-free, transposed stores
            #pragma unroll
            for (int i = 0; i < 2; i++) {
                int e   = t + i * 256;        // 512 qwords
                int kk  = e >> 4;
                int pxc = (e & 15) * 4;
                int p   = pxBase + pxc;
                int n   = p / HW;
                int hw  = p - n * HW;
                bool ok = (p + 3) < P;        // P,HW multiples of 4 here
                int off = (n * Cin + kt + kk) * HW + hw;
                uint2 val = *(const uint2*)(X + (ok ? off : 0));
                if (!ok) { val.x = 0u; val.y = 0u; }
                union { uint2 u2; _Float16 h[4]; } cv; cv.u2 = val;
                #pragma unroll
                for (int j = 0; j < 4; j++) lB[(pxc + j) * 34 + kk] = cv.h[j];
            }
        } else {
            // scalar path (odd HW, or 3x3 im2col), still branch-free loads
            #pragma unroll
            for (int i = 0; i < 8; i++) {
                int idx = t + i * 256;
                int px = idx & 63, kk = idx >> 6;
                int p  = pxBase + px;
                int kg = kt + kk;
                int n  = p / HW;
                int hw = p - n * HW;
                bool ok;
                int off;
                if (KH == 1) {
                    ok  = p < P;
                    off = (n * Cin + kg) * HW + hw;
                } else {
                    int ci = kg / 9, rr = kg - ci * 9;
                    int h0 = hw / W, w0 = hw - h0 * W;
                    int hh = h0 + rr / 3 - 1;
                    int ww = w0 + rr % 3 - 1;
                    ok  = (p < P) && (hh >= 0) && (hh < H) && (ww >= 0) && (ww < W);
                    off = ((n * Cin + ci) * H + hh) * W + ww;
                }
                _Float16 hv = X[ok ? off : 0];
                if (!ok) hv = (_Float16)0.f;
                lB[px * 34 + kk] = hv;
            }
        }

#if HAVE_ASYNC_LDS
        __builtin_amdgcn_s_wait_asynccnt(0);
#endif
        __syncthreads();

        // ---- fragment gathers per ISA 16-bit A(16x32)/B(32x16) lane layouts ----
        union { unsigned int u[8]; v16h h; } af, b0, b1;
        {
            int row = coStrip * 16 + (lane & 15);
            int hk  = (lane >> 4) * 8;
            #pragma unroll
            for (int q = 0; q < 8; q++) {
                int k = ((q & 3) << 1) + hk + ((q >> 2) << 4);
                af.u[q] = *(const unsigned int*)&lA[row * 40 + k];
            }
        }
        {
            int col = pxStrip * 32 + (lane & 15);
            int hk  = (lane >> 4) * 8;
            #pragma unroll
            for (int q = 0; q < 8; q++) {
                int k = ((q & 3) << 1) + hk + ((q >> 2) << 4);
                b0.u[q] = *(const unsigned int*)&lB[col * 34 + k];
                b1.u[q] = *(const unsigned int*)&lB[(col + 16) * 34 + k];
            }
        }
        acc0 = __builtin_amdgcn_wmma_f32_16x16x32_f16(false, af.h, false, b0.h,
                                                      (short)0, acc0, false, false);
        acc1 = __builtin_amdgcn_wmma_f32_16x16x32_f16(false, af.h, false, b1.h,
                                                      (short)0, acc1, false, false);
        __syncthreads();
    }

    // ---- epilogue: C/D layout = lane -> N, VGPR j -> M = j + 8*(lane>=16) ----
    const int half = lane >> 4;
    const int coB  = coBase + coStrip * 16;
    const int px0  = pxBase + pxStrip * 32 + (lane & 15);
    #pragma unroll
    for (int j = 0; j < 8; j++) {
        int co = coB + j + 8 * half;
        if (co >= Cout) continue;
        float s = 1.f, off = 0.f;
        if (use_bn) {
            s = g[co] * rsqrtf(vv[co] + EPS);
            off = b[co] - m[co] * s;
        }
        if (bias) off += bias[co];
        #pragma unroll
        for (int sub = 0; sub < 2; sub++) {
            int p = px0 + sub * 16;
            if (p >= P) continue;
            float val = (sub == 0) ? acc0[j] : acc1[j];
            val = val * s + off;
            if (use_relu) val = fminf(fmaxf(val, 0.f), 6.f);
            int n  = p / HW;
            int hw = p - n * HW;
            int oidx = (n * Cout + co) * HW + hw;
            if (OutF) OutF[oidx] = val;
            else      OutH[oidx] = (_Float16)val;
        }
    }
}

// ---------------------------------------------------------------------------
// Host-side launch
// ---------------------------------------------------------------------------
template <int KH>
static void launch_gemm(const _Float16* X, const _Float16* Wt,
                        const float* g, const float* b, const float* m, const float* v,
                        const float* bias, float* OutF, _Float16* OutH,
                        int N, int Cin, int H, int W, int Cout,
                        int use_bn, int use_relu, hipStream_t stream)
{
    int P = N * H * W;
    dim3 grid((unsigned)((P + 63) / 64), (unsigned)((Cout + 63) / 64));
    conv_gemm_wmma_kernel<KH><<<grid, 256, 0, stream>>>(
        X, Wt, g, b, m, v, bias, OutF, OutH, N, Cin, H, W, Cout, use_bn, use_relu);
}

extern "C" void kernel_launch(void* const* d_in, const int* in_sizes, int n_in,
                              void* d_out, int out_size, void* d_ws, size_t ws_size,
                              hipStream_t stream)
{
    (void)in_sizes; (void)n_in; (void)out_size; (void)ws_size;
    // d_in layout (setup_inputs insertion order):
    //  0: x ; 1: conv1.w, 2-5: conv1.bn {g,b,m,v}
    //  layer i (0..4), base=6+10*i: +0 dw_w, +1..4 dw_bn, +5 pw_w, +6..9 pw_bn
    //  56: head.w1, 57-60: head.bn1, 61: head.w2, 62: head.b2
    #define IN(i) ((const float*)d_in[(i)])

    const int N = 8;

    // ---- workspace carve (all f16): activations ping-pong + converted weights
    _Float16* bufA = (_Float16*)d_ws;                  // 8*32*416*416 = 44,302,336
    _Float16* bufB = bufA + 44302336;                  // 8*32*208*208 = 11,075,584
    _Float16* wf   = bufB + 11075584;

    const int pwWN[5] = {64 * 32, 128 * 64, 256 * 128, 512 * 256, 512 * 512};
    _Float16* pwW[5];
    {
        _Float16* p = wf;
        for (int i = 0; i < 5; i++) { pwW[i] = p; p += pwWN[i]; }
    }
    _Float16* headW1 = pwW[4] + pwWN[4];
    _Float16* headW2 = headW1 + 256 * 4608;

    // convert all GEMM weights to f16 (deterministic, done every launch)
    for (int i = 0; i < 5; i++) {
        int n = pwWN[i];
        f32_to_f16_kernel<<<(n + 255) / 256, 256, 0, stream>>>(IN(6 + 10 * i + 5), pwW[i], n);
    }
    f32_to_f16_kernel<<<(256 * 4608 + 255) / 256, 256, 0, stream>>>(IN(56), headW1, 256 * 4608);
    f32_to_f16_kernel<<<(90 * 256 + 255) / 256, 256, 0, stream>>>(IN(61), headW2, 90 * 256);

    // ---- stem: f32 x -> f16 bufA ----
    {
        int total = N * 416 * 416;
        stem_conv_kernel<<<(unsigned)((total + 255) / 256), 256, 0, stream>>>(
            IN(0), IN(1), IN(2), IN(3), IN(4), IN(5), bufA, N, 416, 416);
    }

    // ---- 5 stages: pool+dw (bufA->bufB), pw GEMM (bufB->bufA) ----
    const int Hs[5]   = {416, 208, 104, 52, 26};
    const int Cin_[5] = {32, 64, 128, 256, 512};
    const int Cou_[5] = {64, 128, 256, 512, 512};
    for (int i = 0; i < 5; i++) {
        const int Lb = 6 + 10 * i;
        const int Hin = Hs[i], Ho = Hin >> 1;
        int total = N * Cin_[i] * Ho * Ho;
        pool_dw_kernel<<<(unsigned)((total + 255) / 256), 256, 0, stream>>>(
            bufA, IN(Lb + 0), IN(Lb + 1), IN(Lb + 2), IN(Lb + 3), IN(Lb + 4),
            bufB, N, Cin_[i], Hin, Hin);
        launch_gemm<1>(bufB, pwW[i], IN(Lb + 6), IN(Lb + 7), IN(Lb + 8), IN(Lb + 9),
                       nullptr, nullptr, bufA, N, Cin_[i], Ho, Ho, Cou_[i],
                       /*bn=*/1, /*relu=*/1, stream);
    }

    // ---- head: 512 -> 256 (3x3 implicit GEMM, K=4608) -> 90 (1x1 + bias, f32 out) ----
    launch_gemm<3>(bufA, headW1, IN(57), IN(58), IN(59), IN(60), nullptr,
                   nullptr, bufB, N, 512, 13, 13, 256, /*bn=*/1, /*relu=*/1, stream);
    launch_gemm<1>(bufB, headW2, nullptr, nullptr, nullptr, nullptr, IN(62),
                   (float*)d_out, nullptr, N, 256, 13, 13, 90, /*bn=*/0, /*relu=*/0, stream);

    #undef IN
}
